// ModelCausality_14302241096268
// MI455X (gfx1250) — matrix operand order repaired and checked by reading, not compiled, not run on hardware
//
#include <hip/hip_runtime.h>
#include <hip/hip_bf16.h>
#include <math.h>
#include <stdint.h>

typedef __attribute__((ext_vector_type(16))) _Float16 v16h;
typedef __attribute__((ext_vector_type(8)))  float    v8f;

#define IN_CH   64
#define HID     256
#define NA      256   // a dimension
#define NB      256   // b dimension
#define NBATCH  8
#define MROWS   (NBATCH * NA)       // 2048 rows for u / v GEMM

#define W1S     264                 // f16 LDS row stride for w1 (16B-aligned, padded)
#define W1_BYTES   (HID * W1S * 2)  // 135168
#define UV_BYTES   (16 * HID * 4)   // 16384 per tile
#define SMEM_BYTES (W1_BYTES + 2 * UV_BYTES)   // 167936 (<320KB WGP LDS)

__device__ __forceinline__ float leaky01(float x) {
    return fmaxf(x, 0.0f) + 0.01f * fminf(x, 0.0f);
}

// Async 16-byte copy global -> LDS (CDNA5 GLOBAL_LOAD_ASYNC_TO_LDS_B128,
// tracked by ASYNCcnt; per-lane LDS byte address in a VGPR).
__device__ __forceinline__ void async_cp16(uint32_t lds_off, const void* gptr) {
    asm volatile("global_load_async_to_lds_b128 %0, %1, off"
                 :: "v"(lds_off), "v"((uint64_t)(uintptr_t)gptr)
                 : "memory");
}
__device__ __forceinline__ void wait_async0() {
    asm volatile("s_wait_asynccnt 0x0" ::: "memory");
}

// -------------------------------------------------------------------------
// Kernel A: convert w1 (256x256 f32) -> f16 row-major in workspace.
// -------------------------------------------------------------------------
__global__ __launch_bounds__(256)
void cvt_w1_kernel(const float* __restrict__ w1, _Float16* __restrict__ w1h)
{
    const int idx  = (blockIdx.x * 256 + threadIdx.x) * 4;   // 65536 elems / 4
    float4 f = *(const float4*)(w1 + idx);
    union { _Float16 h[4]; int2 p; } pk;
    pk.h[0]=(_Float16)f.x; pk.h[1]=(_Float16)f.y;
    pk.h[2]=(_Float16)f.z; pk.h[3]=(_Float16)f.w;
    *(int2*)(w1h + idx) = pk.p;
}

// -------------------------------------------------------------------------
// Kernel 0: u = za @ w0^T + b0 ; v = zb @ w0^T      (M=2048, K=64, N=256)
// One wave per 16-row M-tile; full N (16 n-tiles) in registers.
// -------------------------------------------------------------------------
__global__ __launch_bounds__(32)
void gemm0_kernel(const float* __restrict__ za, const float* __restrict__ zb,
                  const float* __restrict__ w0, const float* __restrict__ b0,
                  float* __restrict__ u, float* __restrict__ v)
{
    const int mat  = blockIdx.x & 1;          // 0 -> u(za), 1 -> v(zb)
    const int m0   = (blockIdx.x >> 1) * 16;
    const int lane = threadIdx.x & 31;
    const int l15  = lane & 15;
    const int hi   = lane >> 4;

    const float* __restrict__ src = mat ? zb : za;
    float*       __restrict__ dst = mat ? v  : u;

    v8f acc[16];
    const v8f vzero = {0.f,0.f,0.f,0.f,0.f,0.f,0.f,0.f};
#pragma unroll
    for (int t = 0; t < 16; ++t) acc[t] = vzero;

    const float* row = src + (m0 + l15) * IN_CH;   // this lane's A row (M = l15)

#pragma unroll
    for (int s = 0; s < 2; ++s) {                  // K = 64 -> 2 steps of 32
        const int klo = 32 * s + 8 * hi;           // A f16 layout: lo run
        const int khi = klo + 16;                  // hi run
        float4 p0 = *(const float4*)(row + klo);
        float4 p1 = *(const float4*)(row + klo + 4);
        float4 p2 = *(const float4*)(row + khi);
        float4 p3 = *(const float4*)(row + khi + 4);
        v16h a;
        a[0]=(_Float16)p0.x; a[1]=(_Float16)p0.y; a[2]=(_Float16)p0.z; a[3]=(_Float16)p0.w;
        a[4]=(_Float16)p1.x; a[5]=(_Float16)p1.y; a[6]=(_Float16)p1.z; a[7]=(_Float16)p1.w;
        a[8]=(_Float16)p2.x; a[9]=(_Float16)p2.y; a[10]=(_Float16)p2.z; a[11]=(_Float16)p2.w;
        a[12]=(_Float16)p3.x; a[13]=(_Float16)p3.y; a[14]=(_Float16)p3.z; a[15]=(_Float16)p3.w;

#pragma unroll
        for (int t = 0; t < 16; ++t) {
            // B[k][n] = w0[n][k]; lane n = 16t + l15, k runs 8hi..+7 and 16+8hi..+7
            const float* brow = w0 + (16 * t + l15) * IN_CH + 32 * s + 8 * hi;
            float4 q0 = *(const float4*)(brow);
            float4 q1 = *(const float4*)(brow + 4);
            float4 q2 = *(const float4*)(brow + 16);
            float4 q3 = *(const float4*)(brow + 20);
            v16h b;
            b[0]=(_Float16)q0.x; b[1]=(_Float16)q0.y; b[2]=(_Float16)q0.z; b[3]=(_Float16)q0.w;
            b[4]=(_Float16)q1.x; b[5]=(_Float16)q1.y; b[6]=(_Float16)q1.z; b[7]=(_Float16)q1.w;
            b[8]=(_Float16)q2.x; b[9]=(_Float16)q2.y; b[10]=(_Float16)q2.z; b[11]=(_Float16)q2.w;
            b[12]=(_Float16)q3.x; b[13]=(_Float16)q3.y; b[14]=(_Float16)q3.z; b[15]=(_Float16)q3.w;

            acc[t] = __builtin_amdgcn_wmma_f32_16x16x32_f16(
                false, a, false, b, (short)0, acc[t], false, false);
        }
    }

    // store D (+b0 folded into u only); D layout: M = r + 8*hi, N = 16t + l15
#pragma unroll
    for (int t = 0; t < 16; ++t) {
        const float bias = mat ? 0.0f : b0[16 * t + l15];
#pragma unroll
        for (int r = 0; r < 8; ++r) {
            dst[(m0 + r + 8 * hi) * HID + 16 * t + l15] = acc[t][r] + bias;
        }
    }
}

// -------------------------------------------------------------------------
// Kernel 1: fused layer1 (K=256 GEMM, w1 in LDS as f16) + layer2 + sigmoid.
// Block: 256 thr (8 waves) covers (n, 16 a's x 16 b's). Wave: 2 row-tiles.
// Row-tile rt: a = a0+rt fixed, rows M <-> b = b0+M.
// All LDS staging via GLOBAL_LOAD_ASYNC_TO_LDS_B128 (ASYNCcnt).
// -------------------------------------------------------------------------
__global__ __launch_bounds__(256)
void fused_kernel(const float* __restrict__ u, const float* __restrict__ v,
                  const _Float16* __restrict__ w1h, const float* __restrict__ b1,
                  const float* __restrict__ w2, const float* __restrict__ b2,
                  float* __restrict__ out)
{
    extern __shared__ __align__(16) char smem[];
    _Float16* __restrict__ w1s = (_Float16*)smem;                 // [HID][W1S] f16
    float*    __restrict__ us  = (float*)(smem + W1_BYTES);       // [16][HID]
    float*    __restrict__ vs  = us + 16 * HID;                   // [16][HID]

    const uint32_t lds_base = (uint32_t)(uintptr_t)(void*)smem;   // LDS byte addr
    const uint32_t lds_us   = lds_base + W1_BYTES;
    const uint32_t lds_vs   = lds_us + UV_BYTES;

    const int n   = blockIdx.z;
    const int a0  = blockIdx.y * 16;
    const int b0v = blockIdx.x * 16;
    const int tid = threadIdx.x;

    // ---- stage w1 (f16, row-major -> padded LDS rows), 16B per async op ----
#pragma unroll
    for (int i = 0; i < 32; ++i) {                 // 8192 chunks / 256 threads
        const int idx = tid + 256 * i;
        const int row = idx >> 5;                  // 0..255
        const int c8  = (idx & 31) << 3;           // 0..248 step 8 halves
        async_cp16(lds_base + (uint32_t)(row * W1S + c8) * 2,
                   w1h + row * HID + c8);
    }
    // ---- stage u/v tiles (f32, exact copies) ----
#pragma unroll
    for (int i = 0; i < 4; ++i) {                  // 1024 chunks / 256 threads
        const int idx = tid + 256 * i;
        const int row = idx >> 6;
        const int c4  = (idx & 63) << 2;
        async_cp16(lds_us + (uint32_t)(row * HID + c4) * 4,
                   u + (n * NA + a0 + row) * HID + c4);
        async_cp16(lds_vs + (uint32_t)(row * HID + c4) * 4,
                   v + (n * NB + b0v + row) * HID + c4);
    }
    wait_async0();
    __syncthreads();

    const int w    = tid >> 5;
    const int lane = tid & 31;
    const int l15  = lane & 15;
    const int hi   = lane >> 4;

#pragma unroll 1
    for (int rt = w; rt < 16; rt += 8) {
        const float* __restrict__ urow = us + rt  * HID;   // same for all rows of tile
        const float* __restrict__ vrow = vs + l15 * HID;   // this lane's row (M=l15)

        v8f acc[16];
        const v8f vzero = {0.f,0.f,0.f,0.f,0.f,0.f,0.f,0.f};
#pragma unroll
        for (int t = 0; t < 16; ++t) acc[t] = vzero;

#pragma unroll 1
        for (int s = 0; s < 8; ++s) {                       // K = 256 -> 8 steps
            const int klo = 32 * s + 8 * hi;
            const int khi = klo + 16;
            // Build A = leaky(u - v) in f16 A-fragment layout
            v16h a;
#pragma unroll
            for (int i = 0; i < 8; ++i) {
                float x = leaky01(urow[klo + i] - vrow[klo + i]);
                float y = leaky01(urow[khi + i] - vrow[khi + i]);
                a[i]     = (_Float16)x;
                a[i + 8] = (_Float16)y;
            }
#pragma unroll
            for (int t = 0; t < 16; ++t) {
                // B[k=h][n=g] = w1[g][h]; lane n = 16t+l15; h runs 8hi..+7 and 16+8hi..+7
                union { int4 q[2]; v16h h; } bu;
                const int4* p = (const int4*)(w1s + (16 * t + l15) * W1S
                                              + 32 * s + 8 * hi);
                bu.q[0] = p[0];
                bu.q[1] = p[2];
                acc[t] = __builtin_amdgcn_wmma_f32_16x16x32_f16(
                    false, a, false, bu.h, (short)0, acc[t], false, false);
            }
        }

        // ---- layer2: bias + leaky, dot with w2, reduce across 16 lanes ----
        float partial[8];
#pragma unroll
        for (int r = 0; r < 8; ++r) partial[r] = 0.0f;

#pragma unroll
        for (int t = 0; t < 16; ++t) {
            const int h = 16 * t + l15;
            const float w2h = w2[h];
            const float b1h = b1[h];
#pragma unroll
            for (int r = 0; r < 8; ++r) {
                partial[r] += leaky01(acc[t][r] + b1h) * w2h;
            }
        }

        const float bb2 = b2[0];
        const int out_row = (n * NA + (a0 + rt)) * NB + b0v;
#pragma unroll
        for (int r = 0; r < 8; ++r) {
            float s = partial[r];
            s += __shfl_xor(s, 1, 32);
            s += __shfl_xor(s, 2, 32);
            s += __shfl_xor(s, 4, 32);
            s += __shfl_xor(s, 8, 32);      // sum over N-lanes within half-wave
            if (l15 == 0) {
                const float x = s + bb2;
                out[out_row + r + 8 * hi] = 1.0f / (1.0f + __expf(-x));
            }
        }
    }
}

// -------------------------------------------------------------------------
extern "C" void kernel_launch(void* const* d_in, const int* in_sizes, int n_in,
                              void* d_out, int out_size, void* d_ws, size_t ws_size,
                              hipStream_t stream)
{
    const float* za = (const float*)d_in[0];
    const float* zb = (const float*)d_in[1];
    const float* w0 = (const float*)d_in[2];
    const float* b0 = (const float*)d_in[3];
    const float* w1 = (const float*)d_in[4];
    const float* b1 = (const float*)d_in[5];
    const float* w2 = (const float*)d_in[6];
    const float* b2 = (const float*)d_in[7];
    float* out = (float*)d_out;

    float*     u   = (float*)d_ws;               // 2048 x 256 f32 = 2 MB
    float*     v   = u + (size_t)MROWS * HID;    // 2 MB more
    _Float16*  w1h = (_Float16*)(v + (size_t)MROWS * HID);   // 128 KB f16

    // w1 f32 -> f16 (one-time, 64 blocks x 256 thr x 4 elems)
    cvt_w1_kernel<<<dim3(64), dim3(256), 0, stream>>>(w1, w1h);

    // u/v precompute: 256 blocks (2 matrices x 128 M-tiles), 1 wave each
    gemm0_kernel<<<dim3(256), dim3(32), 0, stream>>>(za, zb, w0, b0, u, v);

    (void)hipFuncSetAttribute((const void*)fused_kernel,
                              hipFuncAttributeMaxDynamicSharedMemorySize,
                              (int)SMEM_BYTES);

    // fused MLP: (b-tiles, a-tiles, n)
    fused_kernel<<<dim3(NB / 16, NA / 16, NBATCH), dim3(256), SMEM_BYTES, stream>>>(
        u, v, w1h, b1, w2, b2, out);
}
